// SequentialGPTOSSMoE_28887950033622
// MI455X (gfx1250) — compile-verified, matrix operand order checked by reading, not run on hardware
//
#include <hip/hip_runtime.h>
#include <math.h>

// ---------------- types ----------------
typedef __bf16 bf16_t;
typedef __attribute__((ext_vector_type(8)))  bf16_t v8bf;
typedef __attribute__((ext_vector_type(16))) bf16_t v16bf;
typedef __attribute__((ext_vector_type(8)))  float  v8f;

union Frag {
    v16bf v;
    struct { v8bf lo, hi; } p;
};

// ---------------- problem constants ----------------
#define N_TOK 2048
#define HDIM  1024
#define IDIM  1024
#define NEXP  32
#define TOPK  4
#define NSLOT (N_TOK * TOPK)   // 8192 gathered (token,expert) slots
#define ALPHA 1.702f
#define LIMIT 7.0f

// GEMM tiling
#define TM  64
#define TN  64
#define TK  32
#define LDT 40                  // padded LDS K-stride (80B: 16B aligned, bank-conflict free)

__device__ __forceinline__ v8f wmma_bf16(v16bf a, v16bf b, v8f c) {
    // D = A(16x32) * B(32x16) + C, bf16 inputs, f32 accumulate
    return __builtin_amdgcn_wmma_f32_16x16x32_bf16(false, a, false, b, (short)0, c,
                                                   false, false);
}

__device__ __forceinline__ v8f v8f_zero() {
    v8f z;
#pragma unroll
    for (int i = 0; i < 8; ++i) z[i] = 0.0f;
    return z;
}

// pack two f32 -> one dword of two bf16 (lowers to v_cvt_pk_bf16_f32)
__device__ __forceinline__ unsigned pk_bf16(float a, float b) {
    union { bf16_t h[2]; unsigned u; } p;
    p.h[0] = (bf16_t)a;
    p.h[1] = (bf16_t)b;
    return p.u;
}

// ---------------- kernel 0: zero expert counts ----------------
__global__ void moe_zero(int* __restrict__ counts) {
    if (threadIdx.x < NEXP) counts[threadIdx.x] = 0;
}

// ---------------- kernel 1: router (1 wave per token) ----------------
__global__ __launch_bounds__(128) void moe_router(
    const float* __restrict__ x, const float* __restrict__ rw,
    const float* __restrict__ rb, float* __restrict__ scores,
    int* __restrict__ top_idx, float* __restrict__ top_w,
    int* __restrict__ counts)
{
    const int wave = threadIdx.x >> 5;
    const int lane = threadIdx.x & 31;
    const int t = blockIdx.x * 4 + wave;

    // lane e computes logit for expert e
    const float4* x4 = (const float4*)(x + (size_t)t * HDIM);
    const float4* w4 = (const float4*)(rw + (size_t)lane * HDIM);
    float acc = 0.0f;
#pragma unroll 4
    for (int h = 0; h < HDIM / 4; ++h) {
        float4 a = x4[h];
        float4 b = w4[h];
        acc += a.x * b.x + a.y * b.y + a.z * b.z + a.w * b.w;
    }
    float lg = acc + rb[lane];

    // broadcast all 32 logits to every lane
    float vals[NEXP];
#pragma unroll
    for (int j = 0; j < NEXP; ++j) vals[j] = __shfl(lg, j, 32);

    // top-4, ties -> lower index first (matches jax.lax.top_k)
    int bidx[TOPK];
    float bval[TOPK];
    unsigned usedm = 0u;
#pragma unroll
    for (int s = 0; s < TOPK; ++s) {
        float bv = -INFINITY;
        int bi = 0;
#pragma unroll
        for (int j = 0; j < NEXP; ++j) {
            bool free_j = ((usedm >> j) & 1u) == 0u;
            if (free_j && vals[j] > bv) { bv = vals[j]; bi = j; }
        }
        usedm |= (1u << bi);
        bidx[s] = bi;
        bval[s] = bv;
    }

    // softmax over the 4 selected
    float m = bval[0];
    float wv[TOPK];
    float sum = 0.0f;
#pragma unroll
    for (int s = 0; s < TOPK; ++s) { wv[s] = expf(bval[s] - m); sum += wv[s]; }
    float inv = 1.0f / sum;
#pragma unroll
    for (int s = 0; s < TOPK; ++s) wv[s] *= inv;

    // dense scores row
    float sc = 0.0f;
#pragma unroll
    for (int s = 0; s < TOPK; ++s)
        if (bidx[s] == lane) sc = wv[s];
    scores[(size_t)t * NEXP + lane] = sc;

    if (lane < TOPK) {
        top_idx[t * TOPK + lane] = bidx[lane];
        top_w[t * TOPK + lane]   = wv[lane];
        atomicAdd(&counts[bidx[lane]], 1);
    }
}

// ---------------- kernel 2: exclusive scan of counts ----------------
__global__ void moe_scan(const int* __restrict__ counts, int* __restrict__ offsets) {
    if (threadIdx.x == 0) {
        int a = 0;
        for (int e = 0; e < NEXP; ++e) { offsets[e] = a; a += counts[e]; }
        offsets[NEXP] = a;
    }
}

// ---------------- kernel 3: deterministic per-expert compaction ----------------
__global__ __launch_bounds__(256) void moe_compact(
    const int* __restrict__ top_idx, const int* __restrict__ offsets,
    int* __restrict__ list_tok, int* __restrict__ slot_pos)
{
    __shared__ int tmp[256];
    const int e = blockIdx.x;
    const int tid = threadIdx.x;
    const int base = offsets[e];
    int running = 0;
    for (int c = 0; c < NSLOT / 256; ++c) {
        const int i = c * 256 + tid;
        const int mt = (top_idx[i] == e) ? 1 : 0;
        tmp[tid] = mt;
        __syncthreads();
        // Hillis-Steele inclusive scan over 256 entries
        for (int d = 1; d < 256; d <<= 1) {
            int add = (tid >= d) ? tmp[tid - d] : 0;
            __syncthreads();
            tmp[tid] += add;
            __syncthreads();
        }
        const int incl = tmp[tid];
        const int total = tmp[255];
        if (mt) {
            const int g = base + running + (incl - 1);
            list_tok[g] = i >> 2;   // token id
            slot_pos[i] = g;        // gathered position of (token, slot)
        }
        running += total;
        __syncthreads();
    }
}

// ---------------- kernel 4: gathered gate/up GEMM + activation ----------------
__global__ __launch_bounds__(128) void moe_gateup(
    const float* __restrict__ x,
    const float* __restrict__ Wg, const float* __restrict__ bg,
    const float* __restrict__ Wu, const float* __restrict__ bu,
    const int* __restrict__ offsets, const int* __restrict__ list_tok,
    bf16_t* __restrict__ act)
{
    const int e = blockIdx.z;
    const int base = offsets[e];
    const int count = offsets[e + 1] - base;
    const int m0 = blockIdx.y * TM;
    if (m0 >= count) return;
    const int n0 = blockIdx.x * TN;

    __shared__ __align__(16) bf16_t As[TM][LDT];
    __shared__ __align__(16) bf16_t Bgs[TN][LDT];   // transposed: [n][k]
    __shared__ __align__(16) bf16_t Bus[TN][LDT];

    const int tid  = threadIdx.x;
    const int lane = tid & 31;
    const int wv   = tid >> 5;
    const int wm   = (wv & 1) * 32;
    const int wn   = (wv >> 1) * 32;
    const int fr   = lane & 15;           // fragment row/col within 16
    const int kh   = (lane >> 4) * 8;     // K half offset (+ C row offset)

    // A-load mapping: thread covers row ar, 16 K-cols starting at ac0
    const int ar  = tid >> 1;
    const int ac0 = (tid & 1) * 16;
    const int am  = m0 + ar;
    const int atok = (am < count) ? list_tok[base + am] : -1;
    const float* xrow = (atok >= 0) ? (x + (size_t)atok * HDIM) : x;

    // B-load mapping: thread covers K-pair kp/kp+1, 8 n-cols starting at nb
    const int kp = (tid >> 3) * 2;        // 0,2,...,30
    const int nb = (tid & 7) * 8;         // 0..56
    const float* pg = Wg + ((size_t)e * HDIM + kp) * IDIM + n0 + nb;
    const float* pu = Wu + ((size_t)e * HDIM + kp) * IDIM + n0 + nb;

    v8f accg[2][2], accu[2][2];
#pragma unroll
    for (int i = 0; i < 2; ++i)
#pragma unroll
        for (int j = 0; j < 2; ++j) { accg[i][j] = v8f_zero(); accu[i][j] = v8f_zero(); }

    for (int h0 = 0; h0 < HDIM; h0 += TK) {
        // ---- stage A tile (fp32 -> packed bf16, 2x ds_store_b128) ----
        if (atok >= 0) {
            const float4* p = (const float4*)(xrow + h0 + ac0);
            __builtin_prefetch(xrow + h0 + TK + ac0, 0, 3);
            float4 f0 = p[0], f1 = p[1], f2 = p[2], f3 = p[3];
            uint4 o0, o1;
            o0.x = pk_bf16(f0.x, f0.y); o0.y = pk_bf16(f0.z, f0.w);
            o0.z = pk_bf16(f1.x, f1.y); o0.w = pk_bf16(f1.z, f1.w);
            o1.x = pk_bf16(f2.x, f2.y); o1.y = pk_bf16(f2.z, f2.w);
            o1.z = pk_bf16(f3.x, f3.y); o1.w = pk_bf16(f3.z, f3.w);
            *(uint4*)&As[ar][ac0]     = o0;
            *(uint4*)&As[ar][ac0 + 8] = o1;
        } else {
            uint4 z; z.x = z.y = z.z = z.w = 0u;
            *(uint4*)&As[ar][ac0]     = z;
            *(uint4*)&As[ar][ac0 + 8] = z;
        }
        // ---- stage B tiles transposed: K-pair packed ds_store_b32 ----
        {
            const float* g0 = pg + (size_t)h0 * IDIM;
            const float* g1 = g0 + IDIM;
            const float* u0 = pu + (size_t)h0 * IDIM;
            const float* u1 = u0 + IDIM;
            __builtin_prefetch(g0 + (size_t)TK * IDIM, 0, 3);
            __builtin_prefetch(u0 + (size_t)TK * IDIM, 0, 3);
            float4 ga = ((const float4*)g0)[0], gb = ((const float4*)g0)[1];
            float4 gc = ((const float4*)g1)[0], gd = ((const float4*)g1)[1];
            float4 ua = ((const float4*)u0)[0], ub = ((const float4*)u0)[1];
            float4 uc = ((const float4*)u1)[0], ud = ((const float4*)u1)[1];
            float r0[8] = {ga.x, ga.y, ga.z, ga.w, gb.x, gb.y, gb.z, gb.w};
            float r1[8] = {gc.x, gc.y, gc.z, gc.w, gd.x, gd.y, gd.z, gd.w};
            float s0[8] = {ua.x, ua.y, ua.z, ua.w, ub.x, ub.y, ub.z, ub.w};
            float s1[8] = {uc.x, uc.y, uc.z, uc.w, ud.x, ud.y, ud.z, ud.w};
#pragma unroll
            for (int q = 0; q < 8; ++q) {
                *(unsigned*)&Bgs[nb + q][kp] = pk_bf16(r0[q], r1[q]);
                *(unsigned*)&Bus[nb + q][kp] = pk_bf16(s0[q], s1[q]);
            }
        }
        __syncthreads();

        // ---- fragments + WMMA ----
        Frag a[2], fg[2], fu[2];
#pragma unroll
        for (int i = 0; i < 2; ++i) {
            const int rr = wm + i * 16 + fr;
            a[i].p.lo = *(const v8bf*)&As[rr][kh];
            a[i].p.hi = *(const v8bf*)&As[rr][kh + 16];
        }
#pragma unroll
        for (int j = 0; j < 2; ++j) {
            const int cc = wn + j * 16 + fr;
            fg[j].p.lo = *(const v8bf*)&Bgs[cc][kh];
            fg[j].p.hi = *(const v8bf*)&Bgs[cc][kh + 16];
            fu[j].p.lo = *(const v8bf*)&Bus[cc][kh];
            fu[j].p.hi = *(const v8bf*)&Bus[cc][kh + 16];
        }
#pragma unroll
        for (int i = 0; i < 2; ++i)
#pragma unroll
            for (int j = 0; j < 2; ++j) {
                accg[i][j] = wmma_bf16(a[i].v, fg[j].v, accg[i][j]);
                accu[i][j] = wmma_bf16(a[i].v, fu[j].v, accu[i][j]);
            }
        __syncthreads();
    }

    // ---- epilogue: biases + clamp + gated activation, store bf16 act ----
#pragma unroll
    for (int j = 0; j < 2; ++j) {
        const int nI = n0 + wn + j * 16 + fr;
        const float bgv = bg[(size_t)e * IDIM + nI];
        const float buv = bu[(size_t)e * IDIM + nI];
#pragma unroll
        for (int i = 0; i < 2; ++i) {
#pragma unroll
            for (int r = 0; r < 8; ++r) {
                const int ml = wm + i * 16 + r + kh;
                const int mg = m0 + ml;
                if (mg < count) {
                    float g = accg[i][j][r] + bgv;
                    float u = accu[i][j][r] + buv;
                    g = fminf(g, LIMIT);
                    u = fminf(fmaxf(u, -LIMIT), LIMIT);
                    const float sig = 1.0f / (1.0f + __expf(-ALPHA * g));
                    const float a = (u + 1.0f) * (g * sig);
                    act[(size_t)(base + mg) * IDIM + nI] = (bf16_t)a;
                }
            }
        }
    }
}

// ---------------- kernel 5: down-proj GEMM (act @ Wd[e]) ----------------
__global__ __launch_bounds__(128) void moe_down(
    const bf16_t* __restrict__ act, const float* __restrict__ Wd,
    const int* __restrict__ offsets, float* __restrict__ yraw)
{
    const int e = blockIdx.z;
    const int base = offsets[e];
    const int count = offsets[e + 1] - base;
    const int m0 = blockIdx.y * TM;
    if (m0 >= count) return;
    const int n0 = blockIdx.x * TN;   // over H

    __shared__ __align__(16) bf16_t As[TM][LDT];
    __shared__ __align__(16) bf16_t Bds[TN][LDT];

    const int tid  = threadIdx.x;
    const int lane = tid & 31;
    const int wv   = tid >> 5;
    const int wm   = (wv & 1) * 32;
    const int wn   = (wv >> 1) * 32;
    const int fr   = lane & 15;
    const int kh   = (lane >> 4) * 8;

    const int ar  = tid >> 1;
    const int ac0 = (tid & 1) * 16;
    const int am  = m0 + ar;
    const bool avalid = (am < count);
    const bf16_t* arow = act + (size_t)(base + (avalid ? am : 0)) * IDIM;
    // LDS byte offset of this thread's A staging slot (flat addr low 32 bits
    // are the LDS offset within the aperture -- ISA 10.2)
    const unsigned lds_a = (unsigned)(size_t)&As[ar][ac0];

    const int kp = (tid >> 3) * 2;
    const int nb = (tid & 7) * 8;
    const float* pd = Wd + ((size_t)e * IDIM + kp) * HDIM + n0 + nb;

    v8f acc[2][2];
#pragma unroll
    for (int i = 0; i < 2; ++i)
#pragma unroll
        for (int j = 0; j < 2; ++j) acc[i][j] = v8f_zero();

    for (int k0 = 0; k0 < IDIM; k0 += TK) {
        // ---- stage A tile: raw bf16 global->LDS via ASYNC DMA path ----
        if (avalid) {
            const unsigned long long ga =
                (unsigned long long)(size_t)(arow + k0 + ac0);
            asm volatile("global_load_async_to_lds_b128 %0, %1, off"
                         :: "v"(lds_a), "v"(ga) : "memory");
            asm volatile("global_load_async_to_lds_b128 %0, %1, off offset:16"
                         :: "v"(lds_a), "v"(ga) : "memory");
        } else {
            uint4 z; z.x = z.y = z.z = z.w = 0u;
            *(uint4*)&As[ar][ac0]     = z;
            *(uint4*)&As[ar][ac0 + 8] = z;
        }
        // ---- stage Wd tile transposed: K-pair packed ds_store_b32 ----
        {
            const float* d0 = pd + (size_t)k0 * HDIM;
            const float* d1 = d0 + HDIM;
            __builtin_prefetch(d0 + (size_t)TK * HDIM, 0, 3);
            float4 da = ((const float4*)d0)[0], db = ((const float4*)d0)[1];
            float4 dc = ((const float4*)d1)[0], dd = ((const float4*)d1)[1];
            float r0[8] = {da.x, da.y, da.z, da.w, db.x, db.y, db.z, db.w};
            float r1[8] = {dc.x, dc.y, dc.z, dc.w, dd.x, dd.y, dd.z, dd.w};
#pragma unroll
            for (int q = 0; q < 8; ++q)
                *(unsigned*)&Bds[nb + q][kp] = pk_bf16(r0[q], r1[q]);
        }
        // drain async global->LDS copies before the workgroup barrier
        asm volatile("s_wait_asynccnt 0x0" ::: "memory");
        __syncthreads();

        Frag a[2], fb[2];
#pragma unroll
        for (int i = 0; i < 2; ++i) {
            const int rr = wm + i * 16 + fr;
            a[i].p.lo = *(const v8bf*)&As[rr][kh];
            a[i].p.hi = *(const v8bf*)&As[rr][kh + 16];
        }
#pragma unroll
        for (int j = 0; j < 2; ++j) {
            const int cc = wn + j * 16 + fr;
            fb[j].p.lo = *(const v8bf*)&Bds[cc][kh];
            fb[j].p.hi = *(const v8bf*)&Bds[cc][kh + 16];
        }
#pragma unroll
        for (int i = 0; i < 2; ++i)
#pragma unroll
            for (int j = 0; j < 2; ++j)
                acc[i][j] = wmma_bf16(a[i].v, fb[j].v, acc[i][j]);
        __syncthreads();
    }

#pragma unroll
    for (int j = 0; j < 2; ++j) {
        const int nH = n0 + wn + j * 16 + fr;
#pragma unroll
        for (int i = 0; i < 2; ++i) {
#pragma unroll
            for (int r = 0; r < 8; ++r) {
                const int ml = wm + i * 16 + r + kh;
                const int mg = m0 + ml;
                if (mg < count)
                    yraw[(size_t)(base + mg) * HDIM + nH] = acc[i][j][r];
            }
        }
    }
}

// ---------------- kernel 6: weighted gather of 4 expert slots + bias ----------------
__global__ __launch_bounds__(256) void moe_gather(
    const float* __restrict__ yraw, const float* __restrict__ bd,
    const int* __restrict__ top_idx, const float* __restrict__ top_w,
    const int* __restrict__ slot_pos, float* __restrict__ out)
{
    const int gid = blockIdx.x * 256 + threadIdx.x;
    const int t  = gid >> 8;
    const int h0 = (gid & 255) * 4;

    float ax = 0.f, ay = 0.f, az = 0.f, aw = 0.f;
#pragma unroll
    for (int s = 0; s < TOPK; ++s) {
        const int e   = top_idx[t * TOPK + s];
        const float w = top_w[t * TOPK + s];
        const int pos = slot_pos[t * TOPK + s];
        const float4 y = *(const float4*)&yraw[(size_t)pos * HDIM + h0];
        const float4 b = *(const float4*)&bd[(size_t)e * HDIM + h0];
        ax += w * (y.x + b.x);
        ay += w * (y.y + b.y);
        az += w * (y.z + b.z);
        aw += w * (y.w + b.w);
    }
    float4 o; o.x = ax; o.y = ay; o.z = az; o.w = aw;
    *(float4*)&out[(size_t)t * HDIM + h0] = o;
}

// ---------------- launch ----------------
extern "C" void kernel_launch(void* const* d_in, const int* in_sizes, int n_in,
                              void* d_out, int out_size, void* d_ws, size_t ws_size,
                              hipStream_t stream) {
    const float* x  = (const float*)d_in[0];
    const float* rw = (const float*)d_in[1];
    const float* rb = (const float*)d_in[2];
    const float* Wg = (const float*)d_in[3];
    const float* bg = (const float*)d_in[4];
    const float* Wu = (const float*)d_in[5];
    const float* bu = (const float*)d_in[6];
    const float* Wd = (const float*)d_in[7];
    const float* bd = (const float*)d_in[8];
    (void)in_sizes; (void)n_in; (void)out_size;

    float* out    = (float*)d_out;
    float* scores = out + (size_t)N_TOK * HDIM;

    char* ws = (char*)d_ws;
    int*    counts   = (int*)(ws + 0);
    int*    offsets  = (int*)(ws + 256);
    int*    top_idx  = (int*)(ws + 512);
    float*  top_w    = (float*)(ws + 512 + 4 * (size_t)NSLOT);
    int*    list_tok = (int*)(ws + 512 + 8 * (size_t)NSLOT);
    int*    slot_pos = (int*)(ws + 512 + 12 * (size_t)NSLOT);
    bf16_t* act      = (bf16_t*)(ws + 512 + 16 * (size_t)NSLOT);
    float*  yraw     = (float*)(ws + 512 + 16 * (size_t)NSLOT +
                                2 * (size_t)NSLOT * IDIM);
    (void)ws_size;

    moe_zero<<<1, 32, 0, stream>>>(counts);
    moe_router<<<N_TOK / 4, 128, 0, stream>>>(x, rw, rb, scores, top_idx, top_w, counts);
    moe_scan<<<1, 32, 0, stream>>>(counts, offsets);
    moe_compact<<<NEXP, 256, 0, stream>>>(top_idx, offsets, list_tok, slot_pos);
    moe_gateup<<<dim3(IDIM / TN, N_TOK / TM, NEXP), 128, 0, stream>>>(
        x, Wg, bg, Wu, bu, offsets, list_tok, act);
    moe_down<<<dim3(HDIM / TN, N_TOK / TM, NEXP), 128, 0, stream>>>(
        act, Wd, offsets, yraw);
    moe_gather<<<(N_TOK * HDIM / 4) / 256, 256, 0, stream>>>(
        yraw, bd, top_idx, top_w, slot_pos, out);
}
